// ResidualDenseGroup_72292889526313
// MI455X (gfx1250) — matrix operand order repaired and checked by reading, not compiled, not run on hardware
//
#include <hip/hip_runtime.h>
#include <hip/hip_bf16.h>
#include <math.h>

// ---------------------------------------------------------------------------
// CDNA5 (gfx1250, wave32) implementation of the ResidualDenseGroup reference.
// GEMM-shaped work (QKV / proj / fc1 / fc2 / 3x3 convs as implicit GEMM) runs
// on v_wmma_f32_16x16x32_f16.  Each wave owns a 16x32 output tile (two f32
// accumulators share one A fragment).  The 8 waves of a workgroup share one
// 32-column B panel staged once in LDS as f16 (320KB LDS per WGP), so B
// fragments come from two contiguous ds_load_b128 per WMMA instead of 32
// strided VMEM loads, and global weight traffic drops 8x per block.
// ---------------------------------------------------------------------------

typedef __attribute__((ext_vector_type(16))) _Float16 v16h;
typedef __attribute__((ext_vector_type(8)))  float    v8f;

#define ACT_NONE  0
#define ACT_GELU  1
#define ACT_LRELU 2

#define HW   64
#define PIX  4096          // 64*64
#define LDA  192           // channel stride of the dense-concat activation slab

template<int ACT>
__device__ __forceinline__ float apply_act(float v) {
  if (ACT == ACT_GELU)  return 0.5f * v * (1.0f + erff(v * 0.70710678118654752440f));
  if (ACT == ACT_LRELU) return v > 0.0f ? v : 0.2f * v;
  return v;
}

// ---------------------------------------------------------------------------
// WMMA GEMM:  D[M,N] = act( A[M,K] * B[K,N] + bias [+ Res] )
// A row-major (lda), B row-major (ldb).
// Grid: (tilesM/8, N/32).  Block: 256 threads = 8 waves, one 16x32 tile each,
// all sharing blockIdx.y's 32-column B panel staged in LDS (K x 32 f16).
// M % 128 == 0, N % 32 == 0, K % 32 == 0 (holds for every shape here).
// ---------------------------------------------------------------------------
template<int ACT, bool RES>
__global__ void __launch_bounds__(256)
k_gemm_wmma(const float* __restrict__ A, int lda,
            const float* __restrict__ B, int ldb,
            const float* __restrict__ bias,
            const float* __restrict__ Res, int ldr,
            float* __restrict__ D, int ldd,
            int Kdim)
{
  extern __shared__ _Float16 sB[];                  // [n (32)][k (Kdim)] halves

  const int tid  = threadIdx.x;
  const int wid  = tid >> 5;
  const int lane = tid & 31;
  const int tn   = blockIdx.y;
  const int tm   = blockIdx.x * 8 + wid;

  // ---- stage B panel: global (k-major, coalesced along n) -> LDS [n][k] ----
  const int nb   = tn * 32;
  for (int idx = tid; idx < (Kdim << 5); idx += 256) {
    const int k = idx >> 5, n = idx & 31;
    sB[n * Kdim + k] = (_Float16)B[(size_t)k * ldb + nb + n];
  }
  __syncthreads();

  const int half16 = lane >> 4;                     // 0: lanes 0-15, 1: lanes 16-31
  const int l15    = lane & 15;
  const int arow   = tm * 16 + l15;                 // A: this lane's M row
  const int bcol   = nb + l15;                      // B: lane's column in group 0
  const float* Ap0 = A + (size_t)arow * lda;
  const _Float16* sB0 = sB + (size_t)l15 * Kdim;          // column l15
  const _Float16* sB1 = sB + (size_t)(l15 + 16) * Kdim;   // column l15+16

  v8f acc0 = {}, acc1 = {};
  for (int k0 = 0; k0 < Kdim; k0 += 32) {
    v16h a;
    // A layout (matches B): halves 0-15 = K{0..15}+16*half16
    const float* Ap = Ap0 + k0 + (half16 << 4);
#pragma unroll
    for (int i = 0; i < 8; ++i) {
      a[i]     = (_Float16)Ap[i];
      a[i + 8] = (_Float16)Ap[i + 8];
    }
    // B k order used here (same as A): lanes 0-15 hold K0-15, lanes 16-31 hold K16-31
    // -> 16 contiguous halves starting at k0 + half16*16 (32B-aligned ds load)
    const int kb = k0 + (half16 << 4);
    const v16h b0 = *(const v16h*)(sB0 + kb);
    const v16h b1 = *(const v16h*)(sB1 + kb);
    acc0 = __builtin_amdgcn_wmma_f32_16x16x32_f16(false, a, false, b0,
                                                  (short)0, acc0, false, false);
    acc1 = __builtin_amdgcn_wmma_f32_16x16x32_f16(false, a, false, b1,
                                                  (short)0, acc1, false, false);
  }

  // C/D layout: VGPR r, lanes 0-15 -> M = r, lanes 16-31 -> M = r+8
  const float bias0 = bias[bcol], bias1 = bias[bcol + 16];
#pragma unroll
  for (int r = 0; r < 8; ++r) {
    const int m = tm * 16 + r + (half16 << 3);
    float v0 = acc0[r] + bias0;
    float v1 = acc1[r] + bias1;
    if (RES) {
      v0 += Res[(size_t)m * ldr + bcol];
      v1 += Res[(size_t)m * ldr + bcol + 16];
    }
    D[(size_t)m * ldd + bcol]      = apply_act<ACT>(v0);
    D[(size_t)m * ldd + bcol + 16] = apply_act<ACT>(v1);
  }
}

// ---------------------------------------------------------------------------
// 3x3 same-pad conv as implicit GEMM over 9 taps (NHWC activations, OIHW w).
// out[p,o] = act( sum_{ky,kx,c} X[nbr(p,ky,kx),c] * W[o,c,ky,kx] + bias [+Res])
// Grid: (tilesM/8, Cout/32).  The block's 32-out-channel weight slice
// (32 x Cin x 9 <= 108KB f16) is staged once in LDS as [o][tap][c], giving
// contiguous ds_load_b128 B fragments.  Halo handled with branchless selects
// so EXEC stays all-ones around WMMA.
// ---------------------------------------------------------------------------
template<int ACT, bool RES>
__global__ void __launch_bounds__(256)
k_conv3_wmma(const float* __restrict__ X, int ldx,
             const float* __restrict__ W,          // (Cout, Cin, 3, 3)
             const float* __restrict__ bias,
             const float* __restrict__ Res, int ldr,
             float* __restrict__ D, int ldd,
             int Cin)
{
  extern __shared__ _Float16 sW[];                  // [o (32)][tap (9)][c (Cin)]

  const int tid  = threadIdx.x;
  const int wid  = tid >> 5;
  const int lane = tid & 31;
  const int tn   = blockIdx.y;
  const int tm   = blockIdx.x * 8 + wid;

  // ---- stage weight slice: OIHW global -> LDS [o][tap][c] ----
  {
    const int o = tid >> 3, cph = tid & 7;          // 32 o's x 8 c-phases
    const float* Wg = W + (size_t)(tn * 32 + o) * Cin * 9;
    _Float16*    Wl = sW + (size_t)o * 9 * Cin;
    for (int tap = 0; tap < 9; ++tap)
      for (int c = cph; c < Cin; c += 8)
        Wl[tap * Cin + c] = (_Float16)Wg[(size_t)c * 9 + tap];
  }
  __syncthreads();

  const int half16 = lane >> 4;
  const int l15    = lane & 15;
  const int p = tm * 16 + l15;
  const int h = p >> 6, w = p & (HW - 1);
  const int ocol = tn * 32 + l15;
  const _Float16* sW0 = sW + (size_t)l15 * 9 * Cin;         // o = l15
  const _Float16* sW1 = sW + (size_t)(l15 + 16) * 9 * Cin;  // o = l15+16

  v8f acc0 = {}, acc1 = {};
  for (int ky = 0; ky < 3; ++ky)
    for (int kx = 0; kx < 3; ++kx) {
      const int hh = h + ky - 1, ww = w + kx - 1;
      const bool valid = ((unsigned)hh < (unsigned)HW) & ((unsigned)ww < (unsigned)HW);
      const int q = valid ? (hh * HW + ww) : p;       // safe address, zeroed below
      const float* Ap0 = X + (size_t)q * ldx;
      const int tap = ky * 3 + kx;

      for (int k0 = 0; k0 < Cin; k0 += 32) {
        v16h a;
        const float* Ap = Ap0 + k0 + (half16 << 4);
#pragma unroll
        for (int i = 0; i < 8; ++i) {
          a[i]     = (_Float16)(valid ? Ap[i]      : 0.0f);
          a[i + 8] = (_Float16)(valid ? Ap[i + 8]  : 0.0f);
        }
        const int cb = tap * Cin + k0 + (half16 << 4);
        const v16h b0 = *(const v16h*)(sW0 + cb);
        const v16h b1 = *(const v16h*)(sW1 + cb);
        acc0 = __builtin_amdgcn_wmma_f32_16x16x32_f16(false, a, false, b0,
                                                      (short)0, acc0, false, false);
        acc1 = __builtin_amdgcn_wmma_f32_16x16x32_f16(false, a, false, b1,
                                                      (short)0, acc1, false, false);
      }
    }

  const float bias0 = bias[ocol], bias1 = bias[ocol + 16];
#pragma unroll
  for (int r = 0; r < 8; ++r) {
    const int m = tm * 16 + r + (half16 << 3);
    float v0 = acc0[r] + bias0;
    float v1 = acc1[r] + bias1;
    if (RES) {
      v0 += Res[(size_t)m * ldr + ocol];
      v1 += Res[(size_t)m * ldr + ocol + 16];
    }
    D[(size_t)m * ldd + ocol]      = apply_act<ACT>(v0);
    D[(size_t)m * ldd + ocol + 16] = apply_act<ACT>(v1);
  }
}

// ---------------------------------------------------------------------------
// GroupNorm(1,C): single global mean/var over the C x 4096 slab.
// ---------------------------------------------------------------------------
__global__ void __launch_bounds__(256)
k_gn_reduce(const float* __restrict__ slab, int C, float* __restrict__ stats)
{
  __shared__ float s_sum[256], s_sq[256];
  const int tid = threadIdx.x;
  const int total = PIX * C;
  float sum = 0.f, sq = 0.f;
  for (int idx = tid; idx < total; idx += 256) {
    const int pix = idx / C, c = idx - pix * C;
    const float v = slab[(size_t)pix * LDA + c];
    sum += v; sq += v * v;
  }
  s_sum[tid] = sum; s_sq[tid] = sq;
  __syncthreads();
  for (int s = 128; s > 0; s >>= 1) {
    if (tid < s) { s_sum[tid] += s_sum[tid + s]; s_sq[tid] += s_sq[tid + s]; }
    __syncthreads();
  }
  if (tid == 0) {
    const float inv = 1.0f / (float)total;
    const float mu  = s_sum[0] * inv;
    const float var = s_sq[0] * inv - mu * mu;
    stats[0] = mu;
    stats[1] = rsqrtf(var + 1e-5f);
  }
}

__global__ void k_gn_apply(const float* __restrict__ slab, float* __restrict__ g,
                           const float* __restrict__ stats,
                           const float* __restrict__ gw, const float* __restrict__ gb,
                           int C)
{
  const int idx = blockIdx.x * blockDim.x + threadIdx.x;
  if (idx >= PIX * C) return;
  const int pix = idx / C, c = idx - pix * C;
  g[idx] = (slab[(size_t)pix * LDA + c] - stats[0]) * stats[1] * gw[c] + gb[c];
}

// ---------------------------------------------------------------------------
// Per-pixel LayerNorm over C channels.
// ---------------------------------------------------------------------------
__global__ void __launch_bounds__(128)
k_layernorm(const float* __restrict__ A, float* __restrict__ Y,
            const float* __restrict__ lw, const float* __restrict__ lb, int C)
{
  const int p = blockIdx.x * blockDim.x + threadIdx.x;
  if (p >= PIX) return;
  const float* row = A + (size_t)p * C;
  float s = 0.f;
  for (int c = 0; c < C; ++c) s += row[c];
  const float mu = s / (float)C;
  float v = 0.f;
  for (int c = 0; c < C; ++c) { const float d = row[c] - mu; v += d * d; }
  const float is = rsqrtf(v / (float)C + 1e-5f);
  float* out = Y + (size_t)p * C;
  for (int c = 0; c < C; ++c) out[c] = (row[c] - mu) * is * lw[c] + lb[c];
}

// ---------------------------------------------------------------------------
// 7x7 neighborhood attention (NATTEN), 4 heads, head dim d = C/4 (<= 48).
// One thread per (pixel, head); 49 logits staged in LDS; f32 softmax.
// ---------------------------------------------------------------------------
__global__ void __launch_bounds__(128)
k_na2d(const float* __restrict__ qkv, const float* __restrict__ rpb,
       float* __restrict__ out, int C)
{
  const int d = C >> 2;
  const int gid = blockIdx.x * blockDim.x + threadIdx.x;
  if (gid >= PIX * 4) return;
  const int head = gid & 3;
  const int p    = gid >> 2;
  const int h = p >> 6, w = p & (HW - 1);
  int h0 = h - 3; h0 = h0 < 0 ? 0 : (h0 > HW - 7 ? HW - 7 : h0);
  int w0 = w - 3; w0 = w0 < 0 ? 0 : (w0 > HW - 7 ? HW - 7 : w0);

  const int C3 = 3 * C;
  const float* qp = qkv + (size_t)p * C3 + head * d;
  const float  scale = rsqrtf((float)d);

  __shared__ float s_logit[128 * 49];
  float* lg = s_logit + threadIdx.x * 49;

  float mx = -1e30f;
  for (int i = 0; i < 7; ++i)
    for (int j = 0; j < 7; ++j) {
      const int q2 = (h0 + i) * HW + (w0 + j);
      const float* kp = qkv + (size_t)q2 * C3 + C + head * d;
      float s = 0.f;
      for (int c = 0; c < d; ++c) s += qp[c] * kp[c];
      s = s * scale + rpb[head * 169 + (h0 + i - h + 6) * 13 + (w0 + j - w + 6)];
      lg[i * 7 + j] = s;
      mx = fmaxf(mx, s);
    }

  float sum = 0.f;
  for (int t = 0; t < 49; ++t) {
    const float e = expf(lg[t] - mx);
    lg[t] = e; sum += e;
  }
  const float inv = 1.0f / sum;

  float acc[48];
  for (int c = 0; c < d; ++c) acc[c] = 0.f;
  for (int i = 0; i < 7; ++i)
    for (int j = 0; j < 7; ++j) {
      const int q2 = (h0 + i) * HW + (w0 + j);
      const float* vp = qkv + (size_t)q2 * C3 + 2 * C + head * d;
      const float wt = lg[i * 7 + j] * inv;
      for (int c = 0; c < d; ++c) acc[c] += wt * vp[c];
    }
  float* op = out + (size_t)p * C + head * d;
  for (int c = 0; c < d; ++c) op[c] = acc[c];
}

// ---------------------------------------------------------------------------
// Layout shims: NCHW input -> NHWC slab columns 0..63; final y = 0.2*x5 + x.
// ---------------------------------------------------------------------------
__global__ void k_to_nhwc(const float* __restrict__ x, float* __restrict__ slab)
{
  const int idx = blockIdx.x * blockDim.x + threadIdx.x;
  if (idx >= PIX * 64) return;
  const int c = idx >> 12, p = idx & (PIX - 1);
  slab[(size_t)p * LDA + c] = x[idx];
}

__global__ void k_final(const float* __restrict__ x5, const float* __restrict__ x,
                        float* __restrict__ out)
{
  const int idx = blockIdx.x * blockDim.x + threadIdx.x;
  if (idx >= PIX * 64) return;
  const int c = idx >> 12, p = idx & (PIX - 1);
  out[idx] = x5[(size_t)p * 64 + c] * 0.2f + x[idx];
}

// ---------------------------------------------------------------------------
// Host-side orchestration
// ---------------------------------------------------------------------------
template<int ACT, bool RES>
static inline void launch_gemm_t(const float* A, int lda, const float* B, int ldb,
                                 const float* bias, const float* Res, int ldr,
                                 float* D, int ldd, int N, int K,
                                 hipStream_t s)
{
  dim3 grid((PIX >> 4) / 8, N >> 5);
  const size_t shm = (size_t)K * 32 * sizeof(_Float16);
  k_gemm_wmma<ACT, RES><<<grid, 256, shm, s>>>(
      A, lda, B, ldb, bias, Res, ldr, D, ldd, K);
}

template<int ACT, bool RES>
static inline void launch_conv_t(const float* X, int ldx, const float* W,
                                 const float* bias, const float* Res, int ldr,
                                 float* D, int ldd, int Cin, int Cout,
                                 hipStream_t s)
{
  dim3 grid((PIX >> 4) / 8, Cout >> 5);
  const size_t shm = (size_t)32 * 9 * Cin * sizeof(_Float16);
  k_conv3_wmma<ACT, RES><<<grid, 256, shm, s>>>(
      X, ldx, W, bias, Res, ldr, D, ldd, Cin);
}

extern "C" void kernel_launch(void* const* d_in, const int* in_sizes, int n_in,
                              void* d_out, int out_size, void* d_ws, size_t ws_size,
                              hipStream_t stream)
{
  (void)in_sizes; (void)n_in; (void)out_size; (void)ws_size;

  static const int CHANS[5] = {64, 96, 128, 160, 192};
  static const int AOUT[5]  = {32, 32, 32, 32, 64};

  const float* x = (const float*)d_in[0];
  // params[i] dict order: gn_w,gn_b,qkv_w,qkv_b,rpb,proj_w,proj_b,ln_w,ln_b,
  //                       fc1_w,fc1_b,fc2_w,fc2_b,conv_w,conv_b,adj_w,adj_b
  auto P = [&](int blk, int off) -> const float* {
    return (const float*)d_in[1 + blk * 17 + off];
  };

  // workspace carve-up (floats)
  float* ws   = (float*)d_ws;
  float* slab = ws;                      // PIX x 192, dense-concat NHWC activations
  float* g    = slab + (size_t)PIX * 192;
  float* qkv  = g    + (size_t)PIX * 192;   // PIX x 576
  float* obuf = qkv  + (size_t)PIX * 576;
  float* abuf = obuf + (size_t)PIX * 192;
  float* hbuf = abuf + (size_t)PIX * 192;
  float* f1   = hbuf + (size_t)PIX * 192;   // PIX x 768
  float* ybuf = f1   + (size_t)PIX * 768;
  float* rbuf = ybuf + (size_t)PIX * 192;
  float* x5   = rbuf + (size_t)PIX * 192;   // PIX x 64
  float* stats = x5  + (size_t)PIX * 64;    // mean, invstd

  // x (NCHW) -> slab columns [0,64)
  k_to_nhwc<<<(PIX * 64 + 255) / 256, 256, 0, stream>>>(x, slab);

  for (int i = 0; i < 5; ++i) {
    const int C = CHANS[i], O = AOUT[i];
    const int C3 = 3 * C, C4 = 4 * C;

    // ---- GroupNorm(1,C) over the first C slab columns -> g (PIX x C) ----
    k_gn_reduce<<<1, 256, 0, stream>>>(slab, C, stats);
    k_gn_apply<<<(PIX * C + 255) / 256, 256, 0, stream>>>(
        slab, g, stats, P(i, 0), P(i, 1), C);

    // ---- QKV: (PIX,C) x (C,3C) ----
    launch_gemm_t<ACT_NONE, false>(g, C, P(i, 2), C3, P(i, 3), nullptr, 0,
                                   qkv, C3, C3, C, stream);

    // ---- 7x7 neighborhood attention ----
    k_na2d<<<(PIX * 4 + 127) / 128, 128, 0, stream>>>(qkv, P(i, 4), obuf, C);

    // ---- proj + residual (slab cols [0,C)) -> abuf ----
    launch_gemm_t<ACT_NONE, true>(obuf, C, P(i, 5), C, P(i, 6), slab, LDA,
                                  abuf, C, C, C, stream);

    // ---- LayerNorm -> hbuf ----
    k_layernorm<<<(PIX + 127) / 128, 128, 0, stream>>>(abuf, hbuf, P(i, 7), P(i, 8), C);

    // ---- MLP: fc1 + exact GELU, fc2 + residual abuf -> ybuf ----
    launch_gemm_t<ACT_GELU, false>(hbuf, C, P(i, 9), C4, P(i, 10), nullptr, 0,
                                   f1, C4, C4, C, stream);
    launch_gemm_t<ACT_NONE, true>(f1, C4, P(i, 11), C, P(i, 12), abuf, C,
                                  ybuf, C, C, C4, stream);

    // ---- rnab conv3x3 (C->C) + residual slab[:, 0:C) -> rbuf ----
    launch_conv_t<ACT_NONE, true>(ybuf, C, P(i, 13), P(i, 14), slab, LDA,
                                  rbuf, C, C, C, stream);

    // ---- adjust conv3x3 (C->O); blocks 0-3: lrelu into next slab cols ----
    if (i < 4) {
      launch_conv_t<ACT_LRELU, false>(rbuf, C, P(i, 15), P(i, 16), nullptr, 0,
                                      slab + 64 + 32 * i, LDA, C, O, stream);
    } else {
      launch_conv_t<ACT_NONE, false>(rbuf, C, P(i, 15), P(i, 16), nullptr, 0,
                                     x5, 64, C, O, stream);
    }
  }

  // ---- y = 0.2 * x5 + x (NCHW) ----
  k_final<<<(PIX * 64 + 255) / 256, 256, 0, stream>>>(x5, x, (float*)d_out);
}
